// MODEL_6828998001196
// MI455X (gfx1250) — compile-verified, matrix operand order checked by reading, not run on hardware
//
#include <hip/hip_runtime.h>
#include <hip/hip_bf16.h>
#include <stdint.h>

typedef __attribute__((ext_vector_type(16))) _Float16 v16h;
typedef __attribute__((ext_vector_type(8)))  float    v8f;
typedef __attribute__((ext_vector_type(4)))  unsigned int u32x4;
typedef __attribute__((ext_vector_type(8)))  int      i32x8;
typedef __attribute__((ext_vector_type(4)))  int      i32x4;

#define B_    256
#define S_    200
#define M_    32
#define KD_   128
#define VD_   256
#define QD_   128
#define H_    64
#define NROWS (B_*S_)      // 51200
#define XDIM  384          // VD+QD
#define GDIM  256          // 4H

// workspace byte offsets
#define OFF_X     0ull                               // f16 [51200][384]  39.32 MB
#define OFF_G1    39321600ull                        // f32 [51200][256]  52.43 MB
#define OFF_KEYS  91750400ull                        // u64 [51200]        0.41 MB
#define OFF_FIRST 92160000ull                        // u32 [51200]        0.20 MB
#define OFF_WIH16 92364800ull                        // f16 [256][384]
#define OFF_WHH16 92561408ull                        // f16 [256][64]
#define OFF_BIAS  92594176ull                        // f32 [256]  (b_ih+b_hh)

__device__ __forceinline__ float sigf(float x){ return 1.0f/(1.0f+__expf(-x)); }

// K-index base for 16-bit WMMA A(16x32)/B(32x16) fragments (wave32 layout):
// VGPR v holds K pair {kb, kb+1};  lanes 16..31 carry the +8 K-half.
__device__ __forceinline__ int frag_kbase(int v, int hi){
  return (v < 4 ? 2*v : 16 + 2*(v-4)) + (hi ? 8 : 0);
}

// ---------------------------------------------------------------------------
// TDM: issue a 2-D tile load (f16 elements) from global memory into LDS.
//   tile_d0 x tile_d1 elements, row stride stride0 elements, via D# descriptor
//   (ISA cdna5 §8: group0 = {flags, lds_addr, global_addr, type=2},
//    group1 = {data_size=2B, tensor dims, tile dims, dim0 stride}).
// This toolchain exposes the 6-arg builtin:
//   (u32x4 g0, i32x8 g1, i32x4, i32x4, i32x8, i32 cpol)
// ---------------------------------------------------------------------------
#if __has_builtin(__builtin_amdgcn_tensor_load_to_lds)
#define HAVE_TDM 1
__device__ __forceinline__ void tdm_load_tile_f16(const _Float16* gptr, void* ldsptr,
                                                  int tensor_d0, int tensor_d1,
                                                  int tile_d0, int tile_d1,
                                                  int stride0)
{
  uint64_t ga = (uint64_t)(uintptr_t)gptr;
  u32x4 g0;
  g0[0] = 1u;                                          // count=1, user descriptor
  g0[1] = (uint32_t)(uintptr_t)ldsptr;                 // LDS byte address (low 32 bits of flat)
  g0[2] = (uint32_t)(ga & 0xFFFFFFFFu);                // global_addr[31:0]
  g0[3] = (uint32_t)((ga >> 32) & 0x1FFFFFFu) | (2u << 30);  // addr[56:32] | type=2
  i32x8 g1;
  g1[0] = (int)(1u << 16);                             // wg_mask=0, data_size=1 (2B)
  g1[1] = (int)((unsigned)(tensor_d0 & 0xFFFF) << 16); // tensor_dim0[15:0]
  g1[2] = (int)(((unsigned)(tensor_d0 >> 16) & 0xFFFFu) |
                ((unsigned)(tensor_d1 & 0xFFFF) << 16));
  g1[3] = (int)(((unsigned)(tensor_d1 >> 16) & 0xFFFFu) |
                ((unsigned)(tile_d0 & 0xFFFF) << 16)); // tile_dim0
  g1[4] = (int)((unsigned)(tile_d1 & 0xFFFF));         // tile_dim1, tile_dim2=0
  g1[5] = stride0;                                     // tensor_dim0_stride[31:0]
  g1[6] = 0;                                           // stride0 hi, stride1 lo
  g1[7] = 0;
  i32x4 z4 = {0, 0, 0, 0};                             // groups 2/3 unused (2-D tile)
  i32x8 z8 = {0, 0, 0, 0, 0, 0, 0, 0};
  __builtin_amdgcn_tensor_load_to_lds(g0, g1, z4, z4, z8, 0);
}
#else
#define HAVE_TDM 0
#endif

// ---------------------------------------------------------------------------
// Kernel 0: fp32->fp16 weight conversion + fused bias
// ---------------------------------------------------------------------------
__global__ void prep_kernel(const float* __restrict__ Wih, const float* __restrict__ Whh,
                            const float* __restrict__ bih, const float* __restrict__ bhh,
                            _Float16* __restrict__ Wih16, _Float16* __restrict__ Whh16,
                            float* __restrict__ bias)
{
  int i = blockIdx.x * 256 + threadIdx.x;
  if (i < GDIM*XDIM) Wih16[i] = (_Float16)Wih[i];
  if (i < GDIM*H_)   Whh16[i] = (_Float16)Whh[i];
  if (i < GDIM)      bias[i]  = bih[i] + bhh[i];
}

// ---------------------------------------------------------------------------
// Kernel 1: embedding gather + attention softmax + read + trit key pack.
// One block = 32 (b,s) rows.  Writes x=[read|q_emb] as f16, packed trit keys.
// ---------------------------------------------------------------------------
__global__ __launch_bounds__(256)
void attn_kernel(const int* __restrict__ qid,
                 const float* __restrict__ etab,   // [(NQ+1),128]
                 const float* __restrict__ keym,   // [32,128]
                 const float* __restrict__ valm,   // [32,256]
                 _Float16* __restrict__ X,         // [NROWS,384] f16
                 unsigned long long* __restrict__ keys)
{
  __shared__ float s_e[32*128];    // 16 KB embeddings
  __shared__ float s_k[32*128];    // 16 KB key memory
  __shared__ float s_cw[32*32];    //  4 KB logits -> cw
  __shared__ int   s_id[32];
  int t = threadIdx.x;
  int row0 = blockIdx.x * 32;

  if (t < 32) s_id[t] = qid[row0 + t];
  for (int i = t; i < 32*128; i += 256) s_k[i] = keym[i];
  __syncthreads();
  for (int j = 0; j < 16; ++j) {
    int flat = t + 256*j;
    int r = flat >> 7, c = flat & 127;
    s_e[flat] = etab[(size_t)s_id[r]*QD_ + c];
  }
  __syncthreads();

  // scores[r][m] : 1024 dot products of length 128; 4 per thread
  for (int q = 0; q < 4; ++q) {
    int p = t + 256*q;
    int r = p >> 5, m = p & 31;
    float acc = 0.f;
    for (int k = 0; k < 128; ++k) acc += s_e[r*128+k] * s_k[m*128+k];
    s_cw[r*32+m] = acc;
  }
  __syncthreads();

  // softmax over M + triangular-membership trit pack (one thread per row)
  if (t < 32) {
    int r = t;
    float mx = -1e30f;
    for (int m = 0; m < 32; ++m) mx = fmaxf(mx, s_cw[r*32+m]);
    float sum = 0.f;
    for (int m = 0; m < 32; ++m) { float e = __expf(s_cw[r*32+m] - mx); s_cw[r*32+m] = e; sum += e; }
    float inv = 1.0f / sum;
    int kh = 0, kl = 0, p3 = 1;
    for (int m = 0; m < 16; ++m) {
      float cw = s_cw[r*32+m] * inv; s_cw[r*32+m] = cw;
      float w = fminf((cw - 0.075f) / (0.088f - 0.075f), (1.0f - cw) / (1.0f - 0.088f));
      w = fmaxf(w, 0.f);
      int iv = (w >= 0.6f) ? 2 : ((w >= 0.1f) ? 1 : 0);
      kh += iv * p3; p3 *= 3;
    }
    p3 = 1;
    for (int m = 16; m < 32; ++m) {
      float cw = s_cw[r*32+m] * inv; s_cw[r*32+m] = cw;
      float w = fminf((cw - 0.075f) / (0.088f - 0.075f), (1.0f - cw) / (1.0f - 0.088f));
      w = fmaxf(w, 0.f);
      int iv = (w >= 0.6f) ? 2 : ((w >= 0.1f) ? 1 : 0);
      kl += iv * p3; p3 *= 3;
    }
    keys[row0 + r] = ((unsigned long long)(unsigned)kh << 32) | (unsigned)kl;
  }
  __syncthreads();

  // read[r][v] = sum_m cw[r][m]*value[m][v] ; thread t owns column v=t
  float out[32];
  #pragma unroll
  for (int r = 0; r < 32; ++r) out[r] = 0.f;
  for (int m = 0; m < 32; ++m) {
    float vv = valm[m*VD_ + t];
    #pragma unroll
    for (int r = 0; r < 32; ++r) out[r] += s_cw[r*32+m] * vv;
  }
  #pragma unroll
  for (int r = 0; r < 32; ++r)
    X[(size_t)(row0+r)*XDIM + t] = (_Float16)out[r];
  for (int j = 0; j < 16; ++j) {
    int flat = t + 256*j;
    int r = flat >> 7, c = flat & 127;
    X[(size_t)(row0+r)*XDIM + 256 + c] = (_Float16)s_e[flat];
  }
}

// ---------------------------------------------------------------------------
// Kernel 2: G1 = X(f16)[N,384] @ W_ih^T(f16)[384,256] + (b_ih+b_hh), f32 out.
// 128x128 block tile, K staged in 32-chunks via TDM (tensor_load_to_lds),
// v_wmma_f32_16x16x32_f16.  8 waves: wave w owns row-tile w x 8 col tiles.
// ---------------------------------------------------------------------------
__global__ __launch_bounds__(256)
void gate_gemm_kernel(const _Float16* __restrict__ X,
                      const _Float16* __restrict__ W16,   // [256][384]
                      const float* __restrict__ bias,     // [256]
                      float* __restrict__ G1)             // [N][256]
{
  __shared__ _Float16 sX[128*32];   // 8 KB
  __shared__ _Float16 sW[128*32];   // 8 KB  (stored [n][k])
  int tid  = threadIdx.x;
  int lane = tid & 31, wave = tid >> 5;
  int row0 = blockIdx.x * 128;
  int col0 = blockIdx.y * 128;
  int m = lane & 15, hi = lane >> 4;

  v8f acc[8];
  v8f z = {0.f,0.f,0.f,0.f,0.f,0.f,0.f,0.f};
  #pragma unroll
  for (int t = 0; t < 8; ++t) acc[t] = z;

  for (int kc = 0; kc < 12; ++kc) {
    int koff = kc * 32;
    __syncthreads();
#if HAVE_TDM
    // One wave issues two 128x32(halfword) tile DMAs; completion via TENSORcnt,
    // the other waves synchronize on the workgroup barrier.
    if (wave == 0) {
      tdm_load_tile_f16(X   + (size_t)row0*XDIM + koff, sX, XDIM, 128, 32, 128, XDIM);
      tdm_load_tile_f16(W16 + (size_t)col0*XDIM + koff, sW, XDIM, 128, 32, 128, XDIM);
      __builtin_amdgcn_s_wait_tensorcnt(0);
    }
#else
    {
      const uint32_t* Xs = (const uint32_t*)X;
      const uint32_t* Ws = (const uint32_t*)W16;
      uint32_t* dX = (uint32_t*)sX;
      uint32_t* dW = (uint32_t*)sW;
      #pragma unroll
      for (int j = 0; j < 8; ++j) {
        int i = tid + 256*j;
        int r = i >> 4, kd = i & 15;
        dX[i] = Xs[((size_t)(row0+r)*XDIM + koff)/2 + kd];
        dW[i] = Ws[((size_t)(col0+r)*XDIM + koff)/2 + kd];
      }
    }
#endif
    __syncthreads();
    v16h a;
    #pragma unroll
    for (int v = 0; v < 8; ++v) {
      int kb = frag_kbase(v, hi);
      a[2*v]   = sX[(wave*16+m)*32 + kb];
      a[2*v+1] = sX[(wave*16+m)*32 + kb+1];
    }
    #pragma unroll
    for (int t = 0; t < 8; ++t) {
      v16h b;
      #pragma unroll
      for (int v = 0; v < 8; ++v) {
        int kb = frag_kbase(v, hi);
        b[2*v]   = sW[(t*16+m)*32 + kb];
        b[2*v+1] = sW[(t*16+m)*32 + kb+1];
      }
      acc[t] = __builtin_amdgcn_wmma_f32_16x16x32_f16(false, a, false, b,
                                                      (short)0, acc[t], false, false);
    }
  }
  #pragma unroll
  for (int t = 0; t < 8; ++t) {
    int col = col0 + t*16 + m;
    float bv = bias[col];
    #pragma unroll
    for (int v = 0; v < 8; ++v) {
      int r = row0 + wave*16 + v + 8*hi;
      G1[(size_t)r*GDIM + col] = acc[t][v] + bv;
    }
  }
}

// ---------------------------------------------------------------------------
// Kernel 3: LSTM recurrence + fused sigmoid prediction head.
// One wave (32 threads) owns 16 batch rows; W_hh f16 resident in LDS;
// per step: gates = G1 + h@W_hh^T via 2x WMMA per 16-col tile (16 tiles).
// ---------------------------------------------------------------------------
__global__ __launch_bounds__(32)
void lstm_kernel(const float* __restrict__ G1,        // [B*S][256] (row = b*S+s)
                 const _Float16* __restrict__ Whh16,  // [256][64]
                 const float* __restrict__ Wpred,     // [64]
                 const float* __restrict__ bpred,     // [1]
                 const float* __restrict__ init_h,    // [256][64]
                 const float* __restrict__ init_c,    // [256][64]
                 float* __restrict__ pred)            // [B][S]
{
  __shared__ _Float16 sW[GDIM*H_];   // 32 KB
  __shared__ float    sg[16*GDIM];   // 16 KB gate buffer
  __shared__ float    sh[16*H_];     //  4 KB
  __shared__ float    sc[16*H_];     //  4 KB
  __shared__ _Float16 sh16[16*H_];   //  2 KB
  __shared__ float    sWp[H_];
  int lane  = threadIdx.x;
  int bbase = blockIdx.x * 16;
  int m = lane & 15, hi = lane >> 4;

  for (int i = lane; i < GDIM*H_; i += 32) sW[i] = Whh16[i];
  for (int i = lane; i < 16*H_; i += 32) {
    float h0 = init_h[(size_t)(bbase + (i>>6))*H_ + (i&63)];
    float c0 = init_c[(size_t)(bbase + (i>>6))*H_ + (i&63)];
    sh[i] = h0; sc[i] = c0; sh16[i] = (_Float16)h0;
  }
  for (int i = lane; i < H_; i += 32) sWp[i] = Wpred[i];
  float bp = bpred[0];
  __syncthreads();

  for (int s = 0; s < S_; ++s) {
    // A fragments (16x64 h, two K=32 chunks) from LDS h
    v16h a0, a1;
    #pragma unroll
    for (int v = 0; v < 8; ++v) {
      int kb = frag_kbase(v, hi);
      a0[2*v]   = sh16[m*H_ + kb];
      a0[2*v+1] = sh16[m*H_ + kb + 1];
      a1[2*v]   = sh16[m*H_ + 32 + kb];
      a1[2*v+1] = sh16[m*H_ + 32 + kb + 1];
    }
    #pragma unroll 4
    for (int t = 0; t < 16; ++t) {
      v8f acc;
      #pragma unroll
      for (int v = 0; v < 8; ++v)     // seed accumulator with precomputed x@W_ih^T + bias
        acc[v] = G1[((size_t)(bbase + v + 8*hi)*S_ + s)*GDIM + t*16 + m];
      v16h b0, b1;
      #pragma unroll
      for (int v = 0; v < 8; ++v) {
        int kb = frag_kbase(v, hi);
        b0[2*v]   = sW[(t*16+m)*H_ + kb];
        b0[2*v+1] = sW[(t*16+m)*H_ + kb + 1];
        b1[2*v]   = sW[(t*16+m)*H_ + 32 + kb];
        b1[2*v+1] = sW[(t*16+m)*H_ + 32 + kb + 1];
      }
      acc = __builtin_amdgcn_wmma_f32_16x16x32_f16(false, a0, false, b0, (short)0, acc, false, false);
      acc = __builtin_amdgcn_wmma_f32_16x16x32_f16(false, a1, false, b1, (short)0, acc, false, false);
      #pragma unroll
      for (int v = 0; v < 8; ++v)
        sg[(v + 8*hi)*GDIM + t*16 + m] = acc[v];
    }
    __syncthreads();
    // elementwise gates -> (c,h)
    for (int e = 0; e < 32; ++e) {
      int idx = lane + 32*e;
      int r = idx >> 6, hc = idx & 63;
      float gi = sg[r*GDIM + hc];
      float gf = sg[r*GDIM + 64  + hc];
      float gg = sg[r*GDIM + 128 + hc];
      float go = sg[r*GDIM + 192 + hc];
      float c = sigf(gf)*sc[idx] + sigf(gi)*tanhf(gg);
      float h = sigf(go)*tanhf(c);
      sc[idx] = c; sh[idx] = h; sh16[idx] = (_Float16)h;
    }
    __syncthreads();
    if (lane < 16) {                 // fused prediction head
      float acc = bp;
      for (int j = 0; j < 64; ++j) acc += sh[lane*64 + j] * sWp[j];
      pred[(size_t)(bbase + lane)*S_ + s] = sigf(acc);
    }
    __syncthreads();
  }
}

// ---------------------------------------------------------------------------
// Kernels 4a/4b: unique-row rank of packed trit keys.
// rank[i] = |{distinct keys < key[i]}| == lexsort+cumsum result.
// ---------------------------------------------------------------------------
__global__ void firstocc_kernel(const unsigned long long* __restrict__ keys,
                                unsigned int* __restrict__ first)
{
  int i = blockIdx.x*256 + threadIdx.x;
  if (i >= NROWS) return;
  unsigned long long ki = keys[i];
  unsigned int f = 1u;
  for (int j = 0; j < i; ++j)
    if (keys[j] == ki) { f = 0u; break; }
  first[i] = f;
}

__global__ void rank_kernel(const unsigned long long* __restrict__ keys,
                            const unsigned int* __restrict__ first,
                            float* __restrict__ ids_out)
{
  int i = blockIdx.x*256 + threadIdx.x;
  if (i >= NROWS) return;
  unsigned long long ki = keys[i];
  int rank = 0;
  for (int j = 0; j < NROWS; ++j)
    rank += (first[j] && keys[j] < ki) ? 1 : 0;
  ids_out[i] = (float)rank;
}

// ---------------------------------------------------------------------------
extern "C" void kernel_launch(void* const* d_in, const int* in_sizes, int n_in,
                              void* d_out, int out_size, void* d_ws, size_t ws_size,
                              hipStream_t stream) {
  const int*   qid  = (const int*)  d_in[0];
  const float* etab = (const float*)d_in[1];
  const float* keym = (const float*)d_in[2];
  const float* valm = (const float*)d_in[3];
  const float* Wih  = (const float*)d_in[4];
  const float* Whh  = (const float*)d_in[5];
  const float* bih  = (const float*)d_in[6];
  const float* bhh  = (const float*)d_in[7];
  const float* Wp   = (const float*)d_in[8];
  const float* bp   = (const float*)d_in[9];
  const float* ih   = (const float*)d_in[10];
  const float* ic   = (const float*)d_in[11];

  char* ws = (char*)d_ws;
  _Float16*           X     = (_Float16*)(ws + OFF_X);
  float*              G1    = (float*)(ws + OFF_G1);
  unsigned long long* keys  = (unsigned long long*)(ws + OFF_KEYS);
  unsigned int*       first = (unsigned int*)(ws + OFF_FIRST);
  _Float16*           Wih16 = (_Float16*)(ws + OFF_WIH16);
  _Float16*           Whh16 = (_Float16*)(ws + OFF_WHH16);
  float*              bias  = (float*)(ws + OFF_BIAS);

  float* pred = (float*)d_out;            // [B*S]
  float* ids  = (float*)d_out + NROWS;    // [B*S]

  prep_kernel<<<(GDIM*XDIM + 255)/256, 256, 0, stream>>>(Wih, Whh, bih, bhh,
                                                         Wih16, Whh16, bias);
  attn_kernel<<<NROWS/32, 256, 0, stream>>>(qid, etab, keym, valm, X, keys);
  gate_gemm_kernel<<<dim3(NROWS/128, GDIM/128), 256, 0, stream>>>(X, Wih16, bias, G1);
  lstm_kernel<<<B_/16, 32, 0, stream>>>(G1, Whh16, Wp, bp, ih, ic, pred);
  firstocc_kernel<<<(NROWS + 255)/256, 256, 0, stream>>>(keys, first);
  rank_kernel<<<(NROWS + 255)/256, 256, 0, stream>>>(keys, first, ids);
}
